// TransformerEncoderLayer_10204842295356
// MI455X (gfx1250) — matrix operand order repaired and checked by reading, not compiled, not run on hardware
//
#include <hip/hip_runtime.h>
#include <math.h>

typedef __bf16 bf16_t;
typedef __attribute__((ext_vector_type(16))) __bf16 v16bf;
typedef __attribute__((ext_vector_type(8)))  __bf16 v8bf;
typedef __attribute__((ext_vector_type(8)))  float  v8f;

#define N_SEQ 8192
#define D_EMB 256
#define F_FF  1024

union V16U { v16bf v; v8bf h[2]; };

static __device__ __forceinline__ v16bf make16(v8bf lo, v8bf hi) {
    V16U u; u.h[0] = lo; u.h[1] = hi; return u.v;
}

// ---------------------------------------------------------------------------
// fp32 -> bf16 transposed weight conversion: out[n*K + k] = bf16(in[k*N + n])
// ---------------------------------------------------------------------------
__global__ void transpose_bf16_kernel(const float* __restrict__ in,
                                      bf16_t* __restrict__ out, int K, int N) {
    int idx = blockIdx.x * 256 + threadIdx.x;
    if (idx >= K * N) return;
    int k = idx / N, n = idx - k * N;
    out[(size_t)n * K + k] = (bf16_t)in[idx];
}

// ---------------------------------------------------------------------------
// LayerNorm over rows of [8192,256]. Optionally adds pos (q/k path) and
// writes the transposed normalized matrix (V^T for attention).
// ---------------------------------------------------------------------------
__global__ void ln_kernel(const float* __restrict__ x, const float* __restrict__ g,
                          const float* __restrict__ b, const float* __restrict__ pos,
                          bf16_t* __restrict__ out_rm, bf16_t* __restrict__ vt) {
    int row = blockIdx.x, tid = threadIdx.x;
    int lane = tid & 31, wid = tid >> 5;
    __shared__ float red[8];
    float v = x[(size_t)row * D_EMB + tid];
    float s = v;
    for (int o = 16; o; o >>= 1) s += __shfl_xor(s, o, 32);
    if (lane == 0) red[wid] = s;
    __syncthreads();
    float tot = 0.f;
    for (int i = 0; i < 8; i++) tot += red[i];
    float mean = tot * (1.0f / D_EMB);
    __syncthreads();
    float dv = v - mean;
    s = dv * dv;
    for (int o = 16; o; o >>= 1) s += __shfl_xor(s, o, 32);
    if (lane == 0) red[wid] = s;
    __syncthreads();
    tot = 0.f;
    for (int i = 0; i < 8; i++) tot += red[i];
    float var  = tot * (1.0f / D_EMB);
    float nval = dv * rsqrtf(var + 1e-5f) * g[tid] + b[tid];
    float ov = nval;
    if (pos) ov += pos[(size_t)row * D_EMB + tid];
    out_rm[(size_t)row * D_EMB + tid] = (bf16_t)ov;
    if (vt) vt[(size_t)tid * N_SEQ + row] = (bf16_t)nval;
}

// ---------------------------------------------------------------------------
// Flash attention (no 1/sqrt(d) scale, matching reference).
// Wave = 16 query rows. Computes S^T = K @ Q^T (so q == lane&15 in C layout),
// online softmax with per-lane scalars, accumulates O^T = V^T @ P^T.
// ---------------------------------------------------------------------------
__global__ __launch_bounds__(128, 1) void flash_kernel(
        const bf16_t* __restrict__ qk, const bf16_t* __restrict__ vt,
        bf16_t* __restrict__ o_out) {
    int lane = threadIdx.x & 31, wave = threadIdx.x >> 5;
    int hh = lane >> 4, qn = lane & 15;
    int qbase = blockIdx.x * 64 + wave * 16;

    __shared__ __align__(16) bf16_t qs[4][16][D_EMB + 16];
    for (int r = 0; r < 16; r++)
        *(v8bf*)&qs[wave][r][lane * 8] =
            *(const v8bf*)&qk[(size_t)(qbase + r) * D_EMB + lane * 8];
    __syncthreads();

    v8f acc[16];
    for (int t = 0; t < 16; t++)
        for (int r = 0; r < 8; r++) acc[t][r] = 0.f;
    float mrun = -__builtin_inff(), lrun = 0.f;

    for (int kb = 0; kb < N_SEQ / 32; kb++) {
        int key0 = kb * 32;
        v8f st0, st1;
        for (int r = 0; r < 8; r++) { st0[r] = 0.f; st1[r] = 0.f; }
        // S^T tiles: two 16(key)x16(q) C tiles, K-dim = 256 in chunks of 32
        for (int c = 0; c < 8; c++) {
            int koff = c * 32 + hh * 8;            // B(Q^T): {0..7,16..23}/{8..15,24..31}
            v16bf bq = make16(*(const v8bf*)&qs[wave][qn][koff],
                              *(const v8bf*)&qs[wave][qn][koff + 16]);
            int cbase = c * 32 + hh * 8;           // A(K): {0..7,16..23}/{8..15,24..31}
            {
                const bf16_t* kp = qk + (size_t)(key0 + qn) * D_EMB + cbase;
                v16bf ak = make16(*(const v8bf*)kp, *(const v8bf*)(kp + 16));
                st0 = __builtin_amdgcn_wmma_f32_16x16x32_bf16(
                          false, ak, false, bq, (short)0, st0, false, false);
            }
            {
                const bf16_t* kp = qk + (size_t)(key0 + 16 + qn) * D_EMB + cbase;
                v16bf ak = make16(*(const v8bf*)kp, *(const v8bf*)(kp + 16));
                st1 = __builtin_amdgcn_wmma_f32_16x16x32_bf16(
                          false, ak, false, bq, (short)0, st1, false, false);
            }
        }
        // online softmax stats: q == lane&15 -> per-lane scalars
        float bm = st0[0];
        for (int r = 0; r < 8; r++) { bm = fmaxf(bm, st0[r]); bm = fmaxf(bm, st1[r]); }
        bm = fmaxf(bm, __shfl_xor(bm, 16, 32));
        float mnew  = fmaxf(mrun, bm);
        float scale = __expf(mrun - mnew);
        v8f p0, p1; float ps = 0.f;
        for (int r = 0; r < 8; r++) {
            p0[r] = __expf(st0[r] - mnew);
            p1[r] = __expf(st1[r] - mnew);
            ps += p0[r] + p1[r];
        }
        ps += __shfl_xor(ps, 16, 32);
        lrun = lrun * scale + ps;
        mrun = mnew;
        for (int t = 0; t < 16; t++)
            for (int r = 0; r < 8; r++) acc[t][r] *= scale;
        // pack P^T as WMMA B fragment: elem j = P[q=lane&15][key 16h+j]
        v16bf pfrag;
        for (int r = 0; r < 8; r++) {
            float v0l = __shfl(p0[r], qn, 32);
            float v0h = __shfl(p0[r], qn + 16, 32);
            float v1l = __shfl(p1[r], qn, 32);
            float v1h = __shfl(p1[r], qn + 16, 32);
            pfrag[r]     = (bf16_t)(hh ? v1l : v0l);
            pfrag[r + 8] = (bf16_t)(hh ? v1h : v0h);
        }
        // O^T += V^T @ P^T  (A from V^T rows = contiguous key runs)
        for (int dt = 0; dt < 16; dt++) {
            const bf16_t* vp = vt + (size_t)(dt * 16 + qn) * N_SEQ + key0 + hh * 16;
            v16bf av = make16(*(const v8bf*)vp, *(const v8bf*)(vp + 8));
            acc[dt] = __builtin_amdgcn_wmma_f32_16x16x32_bf16(
                          false, av, false, pfrag, (short)0, acc[dt], false, false);
        }
    }
    float inv = 1.0f / lrun;
    int qg = qbase + qn;
    for (int dt = 0; dt < 16; dt++)
        for (int r = 0; r < 8; r++)
            o_out[(size_t)qg * D_EMB + dt * 16 + r + 8 * hh] =
                (bf16_t)(acc[dt][r] * inv);
}

// ---------------------------------------------------------------------------
// Generic bf16 WMMA GEMM: C[M,N] = A[M,K] @ BT[N,K]^T + bias, with epilogue:
//   gelu_mode==0 : outf = acc + bias + res   (fp32, residual)
//   gelu_mode==1 : outb = bf16(gelu_exact(acc + bias))
// Wave computes a 16x128 output strip (8 N-tiles), K chunked by 32.
// ---------------------------------------------------------------------------
__global__ __launch_bounds__(128) void gemm_kernel(
        const bf16_t* __restrict__ A, const bf16_t* __restrict__ BT,
        const float* __restrict__ bias, const float* __restrict__ res,
        float* __restrict__ outf, bf16_t* __restrict__ outb,
        int M, int N, int K, int gelu_mode) {
    int lane = threadIdx.x & 31, wave = threadIdx.x >> 5;
    int hh = lane >> 4, nn = lane & 15;
    int m0 = (blockIdx.x * 4 + wave) * 16;
    int n0 = blockIdx.y * 128;

    v8f acc[8];
    for (int t = 0; t < 8; t++)
        for (int r = 0; r < 8; r++) acc[t][r] = 0.f;

    for (int c = 0; c < K / 32; c++) {
        int cbase = c * 32 + hh * 8;
        const bf16_t* ap = A + (size_t)(m0 + nn) * K + cbase;
        v16bf a = make16(*(const v8bf*)ap, *(const v8bf*)(ap + 16));
        int koff = c * 32 + hh * 8;
        for (int nt = 0; nt < 8; nt++) {
            const bf16_t* bpp = BT + (size_t)(n0 + nt * 16 + nn) * K + koff;
            v16bf bb = make16(*(const v8bf*)bpp, *(const v8bf*)(bpp + 16));
            acc[nt] = __builtin_amdgcn_wmma_f32_16x16x32_bf16(
                          false, a, false, bb, (short)0, acc[nt], false, false);
        }
    }
    for (int nt = 0; nt < 8; nt++) {
        int col = n0 + nt * 16 + nn;
        float bv = bias[col];
        for (int r = 0; r < 8; r++) {
            int row = m0 + r + 8 * hh;
            size_t idx = (size_t)row * N + col;
            float v = acc[nt][r] + bv;
            if (gelu_mode) {
                outb[idx] = (bf16_t)(0.5f * v * (1.0f + erff(v * 0.70710678118f)));
            } else {
                outf[idx] = v + res[idx];
            }
        }
    }
}

// ---------------------------------------------------------------------------
extern "C" void kernel_launch(void* const* d_in, const int* in_sizes, int n_in,
                              void* d_out, int out_size, void* d_ws, size_t ws_size,
                              hipStream_t stream) {
    const float* x   = (const float*)d_in[0];
    const float* pos = (const float*)d_in[1];
    const float* Wp  = (const float*)d_in[2];
    const float* bp  = (const float*)d_in[3];
    const float* W1  = (const float*)d_in[4];
    const float* b1  = (const float*)d_in[5];
    const float* W2  = (const float*)d_in[6];
    const float* b2  = (const float*)d_in[7];
    const float* g1  = (const float*)d_in[8];
    const float* be1 = (const float*)d_in[9];
    const float* g2  = (const float*)d_in[10];
    const float* be2 = (const float*)d_in[11];
    float* out = (float*)d_out;

    char* ws = (char*)d_ws;
    size_t off = 0;
    auto alloc = [&](size_t bytes) {
        char* p = ws + off;
        off += (bytes + 255) & ~(size_t)255;
        return p;
    };
    bf16_t* qk  = (bf16_t*)alloc((size_t)N_SEQ * D_EMB * 2);  // bf16(n + pos), Q and K
    bf16_t* vtm = (bf16_t*)alloc((size_t)D_EMB * N_SEQ * 2);  // V^T = n^T
    bf16_t* ob  = (bf16_t*)alloc((size_t)N_SEQ * D_EMB * 2);  // attention output
    bf16_t* n2  = (bf16_t*)alloc((size_t)N_SEQ * D_EMB * 2);  // LN2 output
    bf16_t* hb  = (bf16_t*)alloc((size_t)N_SEQ * F_FF  * 2);  // gelu(n2@W1+b1)
    float*  x1  = (float*) alloc((size_t)N_SEQ * D_EMB * 4);  // post-MHSA residual
    bf16_t* WpT = (bf16_t*)alloc((size_t)D_EMB * D_EMB * 2);
    bf16_t* W1T = (bf16_t*)alloc((size_t)F_FF  * D_EMB * 2);
    bf16_t* W2T = (bf16_t*)alloc((size_t)D_EMB * F_FF  * 2);

    transpose_bf16_kernel<<<(D_EMB * D_EMB + 255) / 256, 256, 0, stream>>>(Wp, WpT, D_EMB, D_EMB);
    transpose_bf16_kernel<<<(D_EMB * F_FF  + 255) / 256, 256, 0, stream>>>(W1, W1T, D_EMB, F_FF);
    transpose_bf16_kernel<<<(F_FF  * D_EMB + 255) / 256, 256, 0, stream>>>(W2, W2T, F_FF, D_EMB);

    ln_kernel<<<N_SEQ, 256, 0, stream>>>(x, g1, be1, pos, qk, vtm);
    flash_kernel<<<N_SEQ / 64, 128, 0, stream>>>(qk, vtm, ob);
    gemm_kernel<<<dim3(N_SEQ / 64, D_EMB / 128), 128, 0, stream>>>(
        ob, WpT, bp, x, x1, nullptr, N_SEQ, D_EMB, D_EMB, 0);
    ln_kernel<<<N_SEQ, 256, 0, stream>>>(x1, g2, be2, nullptr, n2, nullptr);
    gemm_kernel<<<dim3(N_SEQ / 64, F_FF / 128), 128, 0, stream>>>(
        n2, W1T, b1, nullptr, nullptr, hb, N_SEQ, F_FF, D_EMB, 1);
    gemm_kernel<<<dim3(N_SEQ / 64, D_EMB / 128), 128, 0, stream>>>(
        hb, W2T, b2, x1, out, nullptr, N_SEQ, D_EMB, F_FF, 0);
}
